// BoSsAttention_22385369547245
// MI455X (gfx1250) — matrix operand order repaired and checked by reading, not compiled, not run on hardware
//
#include <hip/hip_runtime.h>
#include <math.h>

// ---------------------------------------------------------------------------
// BoSs attention for MI455X (gfx1250, wave32, WMMA).
// Precision strategy: bf16 WMMA inputs + fp32 accumulation for all matmuls
// (v_wmma_f32_16x16x32_bf16 = 8x the FLOP/inst of the f32 16x16x4 path);
// fp32 for RoPE, masking, online softmax, reductions and final normalize.
// GEMM: one wave computes a 16x128 strip (8 accumulators) so each A-fragment
// is reused across 8 WMMAs -> much better matrix-pipe / VMEM issue balance.
// ---------------------------------------------------------------------------

typedef __attribute__((ext_vector_type(16))) __bf16 v16bf;
typedef __attribute__((ext_vector_type(8)))  float  v8f;

#define WMMA_BF16(a, b, c) \
  __builtin_amdgcn_wmma_f32_16x16x32_bf16(false, (a), false, (b), (short)0, (c), false, false)

static constexpr int kH   = 16;
static constexpr int kKVH = 4;
static constexpr int kD   = 128;
static constexpr int kWIN = 1024;

// ---------------------------------------------------------------------------
// Per-batch sequential scan: rank[l] = (# of t<=l with sid[t]==sid[l]) - 1
// ---------------------------------------------------------------------------
__global__ __launch_bounds__(32) void rank_kernel(const int* __restrict__ sid,
                                                  int* __restrict__ rank, int L) {
  if (threadIdx.x != 0) return;
  const int b = blockIdx.x;
  int cnt[8] = {0, 0, 0, 0, 0, 0, 0, 0};
  const int* s = sid + (size_t)b * L;
  int* r = rank + (size_t)b * L;
  for (int l = 0; l < L; ++l) {
    int st = s[l] & 7;
    r[l] = cnt[st]++;
  }
}

// ---------------------------------------------------------------------------
// C[M,N] = X[M,K] * W[N,K]^T, fp32 in/out, bf16 WMMA inner product.
// One wave per 16x128 C strip: A-frag loaded once per 32-wide k-chunk,
// reused across 8 B-tiles. Requires N % 128 == 0.
// A-frag (16x32 bf16): lane m = lane&15; K = e + (lane>=16?16:0)
// B-frag (32x16 bf16): lane n = lane&15; K = e + (lane>=16?16:0)
// C-frag (16x16 f32):  lane n = lane&15; row = r + (lane>=16?8:0)
// ---------------------------------------------------------------------------
__global__ __launch_bounds__(32) void gemm_xwT_kernel(const float* __restrict__ X,
                                                      const float* __restrict__ W,
                                                      float* __restrict__ C,
                                                      int M, int N, int K) {
  const int m0   = blockIdx.x * 16;
  const int n0   = blockIdx.y * 128;
  const int lane = threadIdx.x & 31;
  const int half = lane >> 4;
  const int lm   = lane & 15;

  v8f acc[8];
#pragma unroll
  for (int t = 0; t < 8; ++t) acc[t] = (v8f){};

  const float* xrow = X + (size_t)(m0 + lm) * K;

  for (int k0 = 0; k0 < K; k0 += 32) {
    // A fragment: loaded once, reused by all 8 N-tiles.
    v16bf a;
#pragma unroll
    for (int e = 0; e < 16; ++e) {
      const int ka = k0 + e + half * 16;
      a[e] = (__bf16)xrow[ka];
    }
#pragma unroll
    for (int t = 0; t < 8; ++t) {
      const float* wrow = W + (size_t)(n0 + t * 16 + lm) * K;
      v16bf b;
#pragma unroll
      for (int e = 0; e < 16; ++e) {
        b[e] = (__bf16)wrow[k0 + e + half * 16];
      }
      acc[t] = WMMA_BF16(a, b, acc[t]);
    }
  }

#pragma unroll
  for (int t = 0; t < 8; ++t) {
#pragma unroll
    for (int r = 0; r < 8; ++r) {
      const int m = m0 + r + half * 8;
      C[(size_t)m * N + n0 + t * 16 + lm] = acc[t][r];
    }
  }
}

// ---------------------------------------------------------------------------
// RoPE on Q (fp32) -> head-major bf16 [B, H, L, D]
// ---------------------------------------------------------------------------
__global__ __launch_bounds__(256) void rope_q_kernel(const float* __restrict__ q,
                                                     const int* __restrict__ pos,
                                                     __bf16* __restrict__ qb,
                                                     int B, int L) {
  const size_t idx   = (size_t)blockIdx.x * blockDim.x + threadIdx.x;
  const size_t total = (size_t)B * L * kH * kD;
  if (idx >= total) return;
  const int d = (int)(idx % kD);
  const int h = (int)((idx / kD) % kH);
  const int l = (int)((idx / ((size_t)kD * kH)) % L);
  const int b = (int)(idx / ((size_t)kD * kH * L));

  const float x  = q[idx];
  const float pr = (d < 64) ? -q[idx + 64] : q[idx - 64];
  const int   dm = d & 63;
  const float inv = powf(10000.0f, -(float)dm * (1.0f / 64.0f));
  const float ang = (float)pos[(size_t)b * L + l] * inv;
  float s, c;
  sincosf(ang, &s, &c);
  const float y = x * c + pr * s;
  qb[(((size_t)(b * kH + h)) * L + l) * kD + d] = (__bf16)y;
}

// ---------------------------------------------------------------------------
// RoPE on K + passthrough V (fp32) -> head-major bf16 [B, KVH, L, D]
// ---------------------------------------------------------------------------
__global__ __launch_bounds__(256) void rope_kv_kernel(const float* __restrict__ k,
                                                      const float* __restrict__ v,
                                                      const int* __restrict__ pos,
                                                      __bf16* __restrict__ kb,
                                                      __bf16* __restrict__ vb,
                                                      int B, int L) {
  const size_t idx   = (size_t)blockIdx.x * blockDim.x + threadIdx.x;
  const size_t total = (size_t)B * L * kKVH * kD;
  if (idx >= total) return;
  const int d = (int)(idx % kD);
  const int h = (int)((idx / kD) % kKVH);
  const int l = (int)((idx / ((size_t)kD * kKVH)) % L);
  const int b = (int)(idx / ((size_t)kD * kKVH * L));

  const float x  = k[idx];
  const float pr = (d < 64) ? -k[idx + 64] : k[idx - 64];
  const int   dm = d & 63;
  const float inv = powf(10000.0f, -(float)dm * (1.0f / 64.0f));
  const float ang = (float)pos[(size_t)b * L + l] * inv;
  float s, c;
  sincosf(ang, &s, &c);
  const float y = x * c + pr * s;
  const size_t dst = (((size_t)(b * kKVH + h)) * L + l) * kD + d;
  kb[dst] = (__bf16)y;
  vb[dst] = (__bf16)v[idx];
}

// ---------------------------------------------------------------------------
// Flash-style masked attention. One wave per (b, h, 16-query tile).
// Key tiles of 32 so P(16x32) is exactly one bf16 A-fragment for P*V.
// out: fp32 [B, L, H*D]
// ---------------------------------------------------------------------------
__global__ __launch_bounds__(32) void attn_kernel(const __bf16* __restrict__ qb,
                                                  const __bf16* __restrict__ kb,
                                                  const __bf16* __restrict__ vb,
                                                  const int* __restrict__ sid,
                                                  const int* __restrict__ rank,
                                                  float* __restrict__ out, int L) {
  const int q0   = blockIdx.x * 16;
  const int h    = blockIdx.y;
  const int b    = blockIdx.z;
  const int kvh  = h >> 2;  // GQA repeat_interleave: head h -> kv head h/4
  const int lane = threadIdx.x & 31;
  const int half = lane >> 4;
  const int lm   = lane & 15;

  const __bf16* Q  = qb + (((size_t)(b * kH + h)) * L + q0) * kD;
  const __bf16* Kp = kb + ((size_t)(b * kKVH + kvh) * L) * kD;
  const __bf16* Vp = vb + ((size_t)(b * kKVH + kvh) * L) * kD;
  const int* sidb  = sid  + (size_t)b * L;
  const int* rnkb  = rank + (size_t)b * L;

  __shared__ float ptile[16 * 32];

  // Resident Q fragments: 4 x (16x32 bf16) covering D=128.
  v16bf qf[4];
#pragma unroll
  for (int c = 0; c < 4; ++c) {
#pragma unroll
    for (int e = 0; e < 16; ++e) {
      const int d = c * 32 + e + half * 16;
      qf[c][e] = Q[(size_t)lm * kD + d];
    }
  }

  // Per-lane per-row (C-frag layout) softmax state + query metadata.
  float rmax[8], rsum[8];
  int sq[8], rq[8];
#pragma unroll
  for (int r = 0; r < 8; ++r) {
    rmax[r] = -1e30f;
    rsum[r] = 0.0f;
    const int m = q0 + r + half * 8;
    sq[r] = sidb[m];
    rq[r] = rnkb[m];
  }

  v8f of[8];
#pragma unroll
  for (int dt = 0; dt < 8; ++dt) of[dt] = (v8f){};

  const float scale = 0.08838834764831845f;  // 1/sqrt(128)
  const int jend = q0 + 15;

  for (int j0 = 0; j0 <= jend; j0 += 32) {
    // ---- S = Q * K^T over D=128 (two 16x16 score frags) ----
    v8f s0 = {}, s1 = {};
#pragma unroll
    for (int c = 0; c < 4; ++c) {
      v16bf b0, b1;
#pragma unroll
      for (int e = 0; e < 16; ++e) {
        const int d = c * 32 + e + half * 16;
        b0[e] = Kp[(size_t)(j0 + lm) * kD + d];
        b1[e] = Kp[(size_t)(j0 + 16 + lm) * kD + d];
      }
      s0 = WMMA_BF16(qf[c], b0, s0);
      s1 = WMMA_BF16(qf[c], b1, s1);
    }

    // ---- mask (same state, causal, rank window) ----
    const int kj0 = j0 + lm, kj1 = kj0 + 16;
    const int sk0 = sidb[kj0], sk1 = sidb[kj1];
    const int rk0 = rnkb[kj0], rk1 = rnkb[kj1];
#pragma unroll
    for (int r = 0; r < 8; ++r) {
      const int m = q0 + r + half * 8;
      const bool ok0 = (sk0 == sq[r]) && (kj0 <= m) && ((rq[r] - rk0) <= kWIN);
      const bool ok1 = (sk1 == sq[r]) && (kj1 <= m) && ((rq[r] - rk1) <= kWIN);
      s0[r] = ok0 ? s0[r] * scale : -1e30f;
      s1[r] = ok1 ? s1[r] * scale : -1e30f;
    }

    // ---- online softmax per row (reduce across the 16-lane half) ----
#pragma unroll
    for (int r = 0; r < 8; ++r) {
      float tm = fmaxf(s0[r], s1[r]);
#pragma unroll
      for (int off = 1; off < 16; off <<= 1) tm = fmaxf(tm, __shfl_xor(tm, off, 32));
      const float nm    = fmaxf(rmax[r], tm);
      const float alpha = __expf(rmax[r] - nm);
      rmax[r] = nm;
      const float p0 = __expf(s0[r] - nm);
      const float p1 = __expf(s1[r] - nm);
      float ts = p0 + p1;
#pragma unroll
      for (int off = 1; off < 16; off <<= 1) ts += __shfl_xor(ts, off, 32);
      rsum[r] = rsum[r] * alpha + ts;
#pragma unroll
      for (int dt = 0; dt < 8; ++dt) of[dt][r] *= alpha;
      const int m = r + half * 8;
      ptile[m * 32 + lm]      = p0;
      ptile[m * 32 + 16 + lm] = p1;
    }
    __syncthreads();

    // ---- P (C layout in LDS) -> bf16 A fragment ----
    v16bf pa;
#pragma unroll
    for (int e = 0; e < 16; ++e) {
      const int kk = e + half * 16;
      pa[e] = (__bf16)ptile[lm * 32 + kk];
    }

    // ---- O += P * V over 8 D-tiles ----
#pragma unroll
    for (int dt = 0; dt < 8; ++dt) {
      v16bf vf;
#pragma unroll
      for (int e = 0; e < 16; ++e) {
        const int j = e + half * 16;
        vf[e] = Vp[(size_t)(j0 + j) * kD + dt * 16 + lm];
      }
      of[dt] = WMMA_BF16(pa, vf, of[dt]);
    }
    __syncthreads();
  }

  // ---- normalize and write out [B, L, H*D] fp32 ----
#pragma unroll
  for (int r = 0; r < 8; ++r) {
    const int m = r + half * 8;
    const float invs = 1.0f / rsum[r];
    const size_t base = ((size_t)(b * L + q0 + m)) * (kH * kD) + (size_t)h * kD;
#pragma unroll
    for (int dt = 0; dt < 8; ++dt) out[base + dt * 16 + lm] = of[dt][r] * invs;
  }
}

// ---------------------------------------------------------------------------
// Host launcher
// ---------------------------------------------------------------------------
extern "C" void kernel_launch(void* const* d_in, const int* in_sizes, int n_in,
                              void* d_out, int out_size, void* d_ws, size_t ws_size,
                              hipStream_t stream) {
  (void)in_sizes; (void)n_in; (void)out_size; (void)ws_size;
  const float* hidden = (const float*)d_in[0];
  const float* Wq     = (const float*)d_in[1];
  const float* Wk     = (const float*)d_in[2];
  const float* Wv     = (const float*)d_in[3];
  const float* Wo     = (const float*)d_in[4];
  const int*   sid    = (const int*)d_in[5];
  const int*   pos    = (const int*)d_in[6];

  const int B = 2, L = 2048, HID = 2048;
  const int M  = B * L;        // 4096
  const int QN = kH * kD;      // 2048
  const int KN = kKVH * kD;    // 512

  char* p = (char*)d_ws;
  float*  q    = (float*)p;  p += (size_t)M * QN * sizeof(float);
  float*  kpr  = (float*)p;  p += (size_t)M * KN * sizeof(float);
  float*  vpr  = (float*)p;  p += (size_t)M * KN * sizeof(float);
  float*  attn = (float*)p;  p += (size_t)M * QN * sizeof(float);
  __bf16* qbf  = (__bf16*)p; p += (size_t)M * QN * sizeof(__bf16);
  __bf16* kbf  = (__bf16*)p; p += (size_t)M * KN * sizeof(__bf16);
  __bf16* vbf  = (__bf16*)p; p += (size_t)M * KN * sizeof(__bf16);
  int*    rnk  = (int*)p;    p += (size_t)M * sizeof(int);

  rank_kernel<<<B, 32, 0, stream>>>(sid, rnk, L);

  // 16x128 strips per wave: grid.y = N/128
  gemm_xwT_kernel<<<dim3(M / 16, QN / 128), 32, 0, stream>>>(hidden, Wq, q,   M, QN, HID);
  gemm_xwT_kernel<<<dim3(M / 16, KN / 128), 32, 0, stream>>>(hidden, Wk, kpr, M, KN, HID);
  gemm_xwT_kernel<<<dim3(M / 16, KN / 128), 32, 0, stream>>>(hidden, Wv, vpr, M, KN, HID);

  {
    const size_t tq = (size_t)M * QN;
    rope_q_kernel<<<(unsigned)((tq + 255) / 256), 256, 0, stream>>>(q, pos, qbf, B, L);
    const size_t tk = (size_t)M * KN;
    rope_kv_kernel<<<(unsigned)((tk + 255) / 256), 256, 0, stream>>>(kpr, vpr, pos, kbf, vbf, B, L);
  }

  attn_kernel<<<dim3(L / 16, kH, B), 32, 0, stream>>>(qbf, kbf, vbf, sid, rnk, attn, L);

  gemm_xwT_kernel<<<dim3(M / 16, HID / 128), 32, 0, stream>>>(attn, Wo, (float*)d_out, M, HID, QN);
}
